// BysMamba_47914655154742
// MI455X (gfx1250) — matrix operand order repaired and checked by reading, not compiled, not run on hardware
//
#include <hip/hip_runtime.h>
#include <hip/hip_bf16.h>
#include <math.h>

// ---------------- problem constants (from reference) ----------------
#define B_SZ    2
#define L_SZ    1024
#define DMODEL  256
#define DSTATE  16
#define DTRANK  16
#define EDIM    512            // EXPAND * D_MODEL
#define DEPTH   8
#define VOCABN  110
#define MROWS   (B_SZ * L_SZ)  // 2048
#define K9      (9 * DMODEL)   // 2304 patch GEMM K

// ---------------- bf16 / WMMA types ----------------
typedef __bf16 bf16;
typedef bf16  v16bf __attribute__((ext_vector_type(16)));
typedef bf16  v8bf  __attribute__((ext_vector_type(8)));
typedef float v8f   __attribute__((ext_vector_type(8)));

__device__ __forceinline__ unsigned short f2b_bits(float f) {
    union { float f; unsigned u; } c; c.f = f;
    unsigned r = c.u + 0x7fffu + ((c.u >> 16) & 1u);   // round-to-nearest-even
    return (unsigned short)(r >> 16);
}

// =====================================================================
// GEMM:  out[M,N] (+)= A[M,K] * W[N,K]^T      (all K multiples of 32)
//   A  : bf16 bits, row-major [M][lda]       (lda = Kpad)
//   Wb : bf16 bits, row-major [Npad64][ldb]  (ldb = Kpad, zero padded)
// Each wave computes a 32x64 strip: 2 A fragments x 4 B fragments ->
// 8x v_wmma_f32_16x16x32_bf16 per 32-K step (1.5 b128 loads per WMMA).
// A order used by this kernel (contiguous K, not the ISA table order): lane<16 -> rows M=lane, elem 0..15 = K
// k0..k0+15; lane>=16 -> k0+16..31 (matches B).
// B 32x16: lane = column, 16 elems = 16 consecutive K (lo lanes k0..15,
// hi lanes k0+16..31).
// epi: 0 = store acc (+bias if non-null)
//      1 = out += scale*acc
//      2 = out = softplus(acc + bias)
// =====================================================================
__global__ void gemm_wmma_kernel(const unsigned short* __restrict__ A, int lda,
                                 const unsigned short* __restrict__ Wb, int ldb,
                                 int K, int M, int N,
                                 float* __restrict__ out, int ldo,
                                 const float* __restrict__ bias,
                                 float scale, int epi)
{
    const int lane  = threadIdx.x & 31;
    const int wave  = threadIdx.x >> 5;
    const int tileM = blockIdx.y * 8 + wave;      // 32-row strips
    if (tileM * 32 >= M) return;

    const int  row0 = tileM * 32;
    const int  col0 = blockIdx.x * 64;
    const int  lm   = lane & 15;
    const bool hi   = (lane >= 16);

    const unsigned short* arow0 = A + (size_t)(row0 + lm) * lda + (hi ? 16 : 0);
    const unsigned short* arow1 = arow0 + (size_t)16 * lda;
    const unsigned short* brow  = Wb + (size_t)(col0 + lm) * ldb + (hi ? 16 : 0);
    const size_t bstep = (size_t)16 * ldb;        // next 16-column tile of W

    v8f acc[2][4];
    #pragma unroll
    for (int g = 0; g < 2; ++g)
        #pragma unroll
        for (int t = 0; t < 4; ++t)
            acc[g][t] = (v8f){0.f,0.f,0.f,0.f,0.f,0.f,0.f,0.f};

    for (int k0 = 0; k0 < K; k0 += 32) {
        v8bf a00 = *reinterpret_cast<const v8bf*>(arow0 + k0);
        v8bf a01 = *reinterpret_cast<const v8bf*>(arow0 + k0 + 8);
        v8bf a10 = *reinterpret_cast<const v8bf*>(arow1 + k0);
        v8bf a11 = *reinterpret_cast<const v8bf*>(arow1 + k0 + 8);
        v16bf af0 = __builtin_shufflevector(a00, a01,
                        0,1,2,3,4,5,6,7,8,9,10,11,12,13,14,15);
        v16bf af1 = __builtin_shufflevector(a10, a11,
                        0,1,2,3,4,5,6,7,8,9,10,11,12,13,14,15);
        v16bf bfr[4];
        #pragma unroll
        for (int t = 0; t < 4; ++t)
            bfr[t] = *reinterpret_cast<const v16bf*>(brow + (size_t)t * bstep + k0);
        #pragma unroll
        for (int t = 0; t < 4; ++t) {
            acc[0][t] = __builtin_amdgcn_wmma_f32_16x16x32_bf16(
                            false, af0, false, bfr[t], (short)0, acc[0][t], false, false);
            acc[1][t] = __builtin_amdgcn_wmma_f32_16x16x32_bf16(
                            false, af1, false, bfr[t], (short)0, acc[1][t], false, false);
        }
    }

    // C/D layout: VGPR r, lane l -> M = r + (l<16?0:8), N = l%16
    #pragma unroll
    for (int g = 0; g < 2; ++g) {
        #pragma unroll
        for (int t = 0; t < 4; ++t) {
            int col = col0 + t * 16 + lm;
            if (col >= N) continue;
            #pragma unroll
            for (int r = 0; r < 8; ++r) {
                int row = row0 + g * 16 + r + (hi ? 8 : 0);
                float a = acc[g][t][r];
                float* op = out + (size_t)row * ldo + col;
                if (epi == 0) {
                    *op = a + (bias ? bias[col] : 0.f);
                } else if (epi == 1) {
                    *op += scale * a;
                } else {
                    float v = a + bias[col];
                    *op = (v > 20.f) ? v : log1pf(__expf(v));
                }
            }
        }
    }
}

// ---------------- weight fp32 -> bf16 (padded [Npad][Kpad]) ----------------
__global__ void wconv_kernel(const float* __restrict__ W,
                             unsigned short* __restrict__ Wb,
                             int N, int K, int Npad, int Kpad)
{
    int idx = blockIdx.x * blockDim.x + threadIdx.x;
    if (idx >= Npad * Kpad) return;
    int n = idx / Kpad, k = idx % Kpad;
    Wb[idx] = (n < N && k < K) ? f2b_bits(W[(size_t)n * K + k]) : (unsigned short)0;
}

// patch_w (O,D,3,3) -> Wb[o][ (m*3+n)*256 + d ] bf16
__global__ void wconv_patch_kernel(const float* __restrict__ pw,
                                   unsigned short* __restrict__ Wb)
{
    int idx = blockIdx.x * blockDim.x + threadIdx.x;
    if (idx >= DMODEL * K9) return;
    int o = idx / K9, k = idx % K9;
    int mn = k / DMODEL, d = k % DMODEL;
    Wb[idx] = f2b_bits(pw[((size_t)o * DMODEL + d) * 9 + mn]);
}

// generic fp32 -> bf16 bits copy
__global__ void cvt_bf16_kernel(const float* __restrict__ src,
                                unsigned short* __restrict__ dst, int n)
{
    int idx = blockIdx.x * blockDim.x + threadIdx.x;
    if (idx < n) dst[idx] = f2b_bits(src[idx]);
}

// dt columns of x_dbl -> padded bf16 [MROWS][32]
__global__ void dtpad_kernel(const float* __restrict__ xdbl,
                             unsigned short* __restrict__ dt_bf)
{
    int idx = blockIdx.x * blockDim.x + threadIdx.x;
    if (idx >= MROWS * 32) return;
    int k = idx & 31, bl = idx >> 5;
    dt_bf[idx] = (k < DTRANK) ? f2b_bits(xdbl[bl * 48 + k]) : (unsigned short)0;
}

// gather x9_bf[bl][(m*3+n)*256 + d] = bf16(embed[tok[bl][mn]][d])
__global__ void gather_x9_kernel(const int* __restrict__ tokens,
                                 const float* __restrict__ embed,
                                 unsigned short* __restrict__ x9)
{
    int idx = blockIdx.x * blockDim.x + threadIdx.x;
    if (idx >= MROWS * K9) return;
    int d  = idx % DMODEL;
    int mn = (idx / DMODEL) % 9;
    int bl = idx / K9;
    int tok = tokens[bl * 9 + mn];
    x9[idx] = f2b_bits(embed[(size_t)tok * DMODEL + d]);
}

// x[bl][d] = 0.5*(embed[center_tok][d] + patch_b[d]);  patch GEMM adds 0.5*acc
__global__ void center_kernel(const int* __restrict__ tokens,
                              const float* __restrict__ embed,
                              const float* __restrict__ patch_b,
                              float* __restrict__ x)
{
    int idx = blockIdx.x * blockDim.x + threadIdx.x;
    if (idx >= MROWS * DMODEL) return;
    int d = idx % DMODEL, bl = idx / DMODEL;
    int tok = tokens[bl * 9 + 4];  // [1][1]
    x[idx] = 0.5f * (embed[(size_t)tok * DMODEL + d] + patch_b[d]);
}

// depthwise causal conv (k=4) + SiLU; dir=0 fwd (l-3+j), dir=1 bwd (l+3-j)
// writes fp32 (for scan) and bf16 (for next GEMM)
__global__ void conv_silu_kernel(const float* __restrict__ xz,    // [MROWS][2*ED]
                                 const float* __restrict__ conv_w,// [ED][1][4]
                                 const float* __restrict__ conv_b,
                                 float* __restrict__ xc,
                                 unsigned short* __restrict__ xc_bf,
                                 int dir)
{
    int idx = blockIdx.x * blockDim.x + threadIdx.x;
    if (idx >= MROWS * EDIM) return;
    int e  = idx % EDIM;
    int bl = idx / EDIM;
    int l  = bl % L_SZ, b = bl / L_SZ;
    float s = conv_b[e];
    #pragma unroll
    for (int j = 0; j < 4; ++j) {
        int lj = (dir == 0) ? (l - 3 + j) : (l + 3 - j);
        if (lj >= 0 && lj < L_SZ)
            s += conv_w[e * 4 + j] * xz[(size_t)(b * L_SZ + lj) * (2 * EDIM) + e];
    }
    float v = s / (1.f + __expf(-s));   // silu
    xc[idx]    = v;
    xc_bf[idx] = f2b_bits(v);
}

// sequential selective scan; one thread per (b, channel e); 1024 steps
__global__ void scan_kernel(const float* __restrict__ xz,    // z at +EDIM
                            const float* __restrict__ xc,
                            const float* __restrict__ x_dbl, // [MROWS][48]; Bs@16, Cs@32
                            const float* __restrict__ delta,
                            const float* __restrict__ A_log, // [ED][16]
                            const float* __restrict__ Dp,
                            unsigned short* __restrict__ y_bf, int dir)
{
    int e = blockIdx.x * blockDim.x + threadIdx.x;  // grid.x*256 covers ED
    int b = blockIdx.y;
    float Arow[DSTATE];
    #pragma unroll
    for (int n = 0; n < DSTATE; ++n) Arow[n] = -__expf(A_log[e * DSTATE + n]);
    float Dv = Dp[e];
    float h[DSTATE];
    #pragma unroll
    for (int n = 0; n < DSTATE; ++n) h[n] = 0.f;

    __shared__ float sBC[32];
    for (int step = 0; step < L_SZ; ++step) {
        int l = dir ? (L_SZ - 1 - step) : step;
        size_t bl = (size_t)b * L_SZ + l;
        if (threadIdx.x < 32) sBC[threadIdx.x] = x_dbl[bl * 48 + 16 + threadIdx.x];
        __syncthreads();
        float d   = delta[bl * EDIM + e];
        float xcv = xc[bl * EDIM + e];
        float zv  = xz[bl * (2 * EDIM) + EDIM + e];
        float yv = 0.f;
        #pragma unroll
        for (int n = 0; n < DSTATE; ++n) {
            float dA = __expf(d * Arow[n]);
            h[n] = dA * h[n] + d * sBC[n] * xcv;
            yv  += h[n] * sBC[16 + n];
        }
        yv += Dv * xcv;
        yv *= zv / (1.f + __expf(-zv));
        y_bf[bl * EDIM + e] = f2b_bits(yv);
        __syncthreads();
    }
}

// =====================================================================
// Host-side orchestration
// =====================================================================
namespace {

struct MambaP {
    const float *in_w, *conv_w, *conv_b, *x_w, *dt_w, *dt_b, *A_log, *D, *out_w;
};

struct Scratch {
    float *x, *xz, *xc, *xdbl, *delta;
    unsigned short *xbf, *x9bf, *xcbf, *dtbf, *ybf;
    unsigned short *wpatch, *whead, *w_in, *w_x, *w_dt, *w_out;
};

inline int rup(int v, int a) { return (v + a - 1) / a * a; }

void launch_gemm(hipStream_t s, const unsigned short* A, int lda,
                 const unsigned short* Wb, int ldb, int K, int M, int N,
                 float* out, int ldo, const float* bias, float scale, int epi)
{
    dim3 grid((N + 63) / 64, ((M + 31) / 32 + 7) / 8);
    gemm_wmma_kernel<<<grid, 256, 0, s>>>(A, lda, Wb, ldb, K, M, N,
                                          out, ldo, bias, scale, epi);
}

void launch_wconv(hipStream_t s, const float* W, unsigned short* Wb, int N, int K)
{
    int Npad = rup(N, 64), Kpad = rup(K, 32);
    int elems = Npad * Kpad;
    wconv_kernel<<<(elems + 255) / 256, 256, 0, s>>>(W, Wb, N, K, Npad, Kpad);
}

void launch_cvt(hipStream_t s, const float* src, unsigned short* dst, int n)
{
    cvt_bf16_kernel<<<(n + 255) / 256, 256, 0, s>>>(src, dst, n);
}

// x_acc += scale * mamba(A_bf)   (direction dir); A_bf = bf16 input [M][256]
void run_mamba(hipStream_t s, const Scratch& w, const unsigned short* A_bf,
               float* x_acc, float scale, int dir, const MambaP& p)
{
    // weight conversions (every call; deterministic, no caching)
    launch_wconv(s, p.in_w,  w.w_in,  2 * EDIM, DMODEL);                 // (1024,256)
    launch_wconv(s, p.x_w,   w.w_x,   DTRANK + 2 * DSTATE, EDIM);        // (48,512) -> 64 rows
    launch_wconv(s, p.dt_w,  w.w_dt,  EDIM, DTRANK);                     // (512,16) -> Kpad 32
    launch_wconv(s, p.out_w, w.w_out, DMODEL, EDIM);                     // (256,512)

    // xz = x @ in_w^T
    launch_gemm(s, A_bf, DMODEL, w.w_in, DMODEL, DMODEL, MROWS, 2 * EDIM,
                w.xz, 2 * EDIM, nullptr, 0.f, 0);
    // depthwise conv + silu
    conv_silu_kernel<<<(MROWS * EDIM + 255) / 256, 256, 0, s>>>(
        w.xz, p.conv_w, p.conv_b, w.xc, w.xcbf, dir);
    // x_dbl = xc @ x_w^T  (N=48: dt|Bs|Cs)
    launch_gemm(s, w.xcbf, EDIM, w.w_x, EDIM, EDIM, MROWS, DTRANK + 2 * DSTATE,
                w.xdbl, 48, nullptr, 0.f, 0);
    // dt -> padded bf16
    dtpad_kernel<<<(MROWS * 32 + 255) / 256, 256, 0, s>>>(w.xdbl, w.dtbf);
    // delta = softplus(dt @ dt_w^T + dt_b)
    launch_gemm(s, w.dtbf, 32, w.w_dt, 32, 32, MROWS, EDIM,
                w.delta, EDIM, p.dt_b, 0.f, 2);
    // selective scan -> y (bf16)
    scan_kernel<<<dim3(EDIM / 256, B_SZ), 256, 0, s>>>(
        w.xz, w.xc, w.xdbl, w.delta, p.A_log, p.D, w.ybf, dir);
    // x_acc += scale * (y @ out_w^T)
    launch_gemm(s, w.ybf, EDIM, w.w_out, EDIM, EDIM, MROWS, DMODEL,
                x_acc, DMODEL, nullptr, scale, 1);
}

} // namespace

extern "C" void kernel_launch(void* const* d_in, const int* in_sizes, int n_in,
                              void* d_out, int out_size, void* d_ws, size_t ws_size,
                              hipStream_t stream)
{
    (void)in_sizes; (void)n_in; (void)out_size; (void)ws_size;

    const int*   tokens  = (const int*)  d_in[0];
    const float* embed   = (const float*)d_in[1];
    const float* patch_w = (const float*)d_in[2];
    const float* patch_b = (const float*)d_in[3];
    const float* head_w  = (const float*)d_in[31];

    auto getP = [&](int base, int layer) -> MambaP {
        MambaP p;
        p.in_w   = (const float*)d_in[base + 0] + (size_t)layer * 2 * EDIM * DMODEL;
        p.conv_w = (const float*)d_in[base + 1] + (size_t)layer * EDIM * 4;
        p.conv_b = (const float*)d_in[base + 2] + (size_t)layer * EDIM;
        p.x_w    = (const float*)d_in[base + 3] + (size_t)layer * (DTRANK + 2 * DSTATE) * EDIM;
        p.dt_w   = (const float*)d_in[base + 4] + (size_t)layer * EDIM * DTRANK;
        p.dt_b   = (const float*)d_in[base + 5] + (size_t)layer * EDIM;
        p.A_log  = (const float*)d_in[base + 6] + (size_t)layer * EDIM * DSTATE;
        p.D      = (const float*)d_in[base + 7] + (size_t)layer * EDIM;
        p.out_w  = (const float*)d_in[base + 8] + (size_t)layer * DMODEL * EDIM;
        return p;
    };
    MambaP in_p  = getP(4, 0);
    MambaP out_p = getP(22, 0);

    // ---- workspace bump allocator ----
    char* wp = (char*)d_ws;
    auto alloc = [&](size_t bytes) -> void* {
        void* r = wp; wp += (bytes + 255) / 256 * 256; return r;
    };
    Scratch w;
    w.x      = (float*)alloc((size_t)MROWS * DMODEL * 4);
    w.xz     = (float*)alloc((size_t)MROWS * 2 * EDIM * 4);
    w.xc     = (float*)alloc((size_t)MROWS * EDIM * 4);
    w.xdbl   = (float*)alloc((size_t)MROWS * 48 * 4);
    w.delta  = (float*)alloc((size_t)MROWS * EDIM * 4);
    w.xbf    = (unsigned short*)alloc((size_t)MROWS * DMODEL * 2);   // frozen bf16 input
    w.x9bf   = (unsigned short*)alloc((size_t)MROWS * K9 * 2);
    w.xcbf   = (unsigned short*)alloc((size_t)MROWS * EDIM * 2);
    w.dtbf   = (unsigned short*)alloc((size_t)MROWS * 32 * 2);
    w.ybf    = (unsigned short*)alloc((size_t)MROWS * EDIM * 2);
    w.wpatch = (unsigned short*)alloc((size_t)DMODEL * K9 * 2);
    w.whead  = (unsigned short*)alloc((size_t)rup(VOCABN, 64) * DMODEL * 2);
    w.w_in   = (unsigned short*)alloc((size_t)2 * EDIM * DMODEL * 2);
    w.w_x    = (unsigned short*)alloc((size_t)64 * EDIM * 2);
    w.w_dt   = (unsigned short*)alloc((size_t)EDIM * 32 * 2);
    w.w_out  = (unsigned short*)alloc((size_t)DMODEL * EDIM * 2);

    const int xn = MROWS * DMODEL;

    // ---- 1. embed + patch conv:  x = (center + patch + bias)/2 ----
    gather_x9_kernel<<<(MROWS * K9 + 255) / 256, 256, 0, stream>>>(tokens, embed, w.x9bf);
    center_kernel<<<(xn + 255) / 256, 256, 0, stream>>>(tokens, embed, patch_b, w.x);
    wconv_patch_kernel<<<(DMODEL * K9 + 255) / 256, 256, 0, stream>>>(patch_w, w.wpatch);
    launch_gemm(stream, w.x9bf, K9, w.wpatch, K9, K9, MROWS, DMODEL,
                w.x, DMODEL, nullptr, 0.5f, 1);

    // ---- 2. x += bidir(x, in_p) ----  (bf16 snapshot = frozen input)
    launch_cvt(stream, w.x, w.xbf, xn);
    run_mamba(stream, w, w.xbf, w.x, 0.5f, 0, in_p);
    run_mamba(stream, w, w.xbf, w.x, 0.5f, 1, in_p);

    // ---- 3. 8 residual layers ----
    for (int i = 0; i < DEPTH; ++i) {
        MambaP lp = getP(13, i);
        launch_cvt(stream, w.x, w.xbf, xn);
        run_mamba(stream, w, w.xbf, w.x, 1.0f, 0, lp);
    }

    // ---- 4. x += bidir(x, out_p) ----
    launch_cvt(stream, w.x, w.xbf, xn);
    run_mamba(stream, w, w.xbf, w.x, 0.5f, 0, out_p);
    run_mamba(stream, w, w.xbf, w.x, 0.5f, 1, out_p);

    // ---- 5. head: logits = x @ head_w^T ----
    launch_cvt(stream, w.x, w.xbf, xn);
    launch_wconv(stream, head_w, w.whead, VOCABN, DMODEL);
    launch_gemm(stream, w.xbf, DMODEL, w.whead, DMODEL, DMODEL, MROWS, VOCABN,
                (float*)d_out, VOCABN, nullptr, 0.f, 0);
}
